// Mixer_32512902430854
// MI455X (gfx1250) — matrix operand order repaired and checked by reading, not compiled, not run on hardware
//
#include <hip/hip_runtime.h>
#include <hip/hip_bf16.h>

typedef __attribute__((ext_vector_type(2)))  float    v2f;
typedef __attribute__((ext_vector_type(8)))  float    v8f;
typedef __attribute__((ext_vector_type(16))) _Float16 v16h;
typedef __attribute__((ext_vector_type(8)))  _Float16 v8h;
typedef __attribute__((ext_vector_type(4)))  _Float16 h4;

#define BB 1024
#define NN 32
#define DD 512
#define HH 1024          // 2*D
#define ZS 516           // k1 LDS stride for z rows (floats)
#define XS2 520          // k2 LDS stride for X tile (halfs): bytes/4 == 4 mod 64
#define HS2 1032         // k2 LDS stride for H tile (halfs): bytes/4 == 4 mod 64

// ---------------------------------------------------------------------------
// Kernel 0: convert W (fp32, K-major) -> f16, N-major (transposed per expert)
// dst[e][c][r] = (f16) src[e][r][c]
// ---------------------------------------------------------------------------
__global__ __launch_bounds__(256) void k0_transpose_f16(
    const float* __restrict__ src, _Float16* __restrict__ dst, int R, int C)
{
    __shared__ float tile[32][33];
    const int e  = blockIdx.z;
    const int r0 = blockIdx.y << 5;
    const int c0 = blockIdx.x << 5;
    const int tx = threadIdx.x & 31;
    const int ty = threadIdx.x >> 5;
    const float* s = src + (size_t)e * R * C;
    _Float16*    d = dst + (size_t)e * R * C;
#pragma unroll
    for (int i = 0; i < 4; ++i) {
        int r = ty + i * 8;
        tile[r][tx] = s[(size_t)(r0 + r) * C + c0 + tx];
    }
    __syncthreads();
#pragma unroll
    for (int i = 0; i < 4; ++i) {
        int cc = ty + i * 8;
        d[(size_t)(c0 + cc) * R + r0 + tx] = (_Float16)tile[tx][cc];
    }
}

// ---------------------------------------------------------------------------
// Kernel 1: per-batch  Az = A_sub^T @ z[b] (fp32 VALU), LayerNorm (fp32),
// scatter rows into expert-grouped f16 buffer Xh[e][b][:], record nsrc[e][b].
// ---------------------------------------------------------------------------
__global__ __launch_bounds__(256) void k1_az_ln_scatter(
    const float* __restrict__ z, const int* __restrict__ idx,
    const float* __restrict__ A, const float* __restrict__ gamma,
    const float* __restrict__ beta, _Float16* __restrict__ Xh,
    int* __restrict__ nsrc)
{
    extern __shared__ char dynsmem[];
    float* zl    = (float*)dynsmem;      // 32 * 516
    float* asub  = zl + 32 * ZS;         // 32 * 33
    float* sred  = asub + 32 * 33;       // 256
    float* ssred = sred + 256;           // 256
    __shared__ int sidx[32];

    const int b   = blockIdx.x;
    const int tid = threadIdx.x;

    if (tid < 32) {
        int e = idx[b * NN + tid];
        sidx[tid] = e;
        nsrc[e * BB + b] = tid;          // n such that idx[b,n] == e
    }
    __syncthreads();

    for (int u = tid; u < 32 * 32; u += 256) {
        int k = u >> 5, j = u & 31;
        asub[k * 33 + j] = A[sidx[k] * NN + sidx[j]];
    }
    {
        const float4* zsrc = (const float4*)(z + (size_t)b * NN * DD);
        for (int u = tid; u < NN * DD / 4; u += 256) {
            int row = u >> 7, c4 = u & 127;
            *(float4*)(zl + row * ZS + c4 * 4) = zsrc[u];
        }
    }
    __syncthreads();

    // j varies across lanes; dg uniform per wave -> z reads are broadcasts
    const int j  = tid & 31;
    const int dg = tid >> 5;
    const int d0 = dg * 64;

    float4 acc[16];
#pragma unroll
    for (int i = 0; i < 16; ++i) acc[i] = make_float4(0.f, 0.f, 0.f, 0.f);

    for (int k = 0; k < 32; ++k) {
        float a = asub[k * 33 + j];
        const float4* zr = (const float4*)(zl + k * ZS + d0);
#pragma unroll
        for (int i = 0; i < 16; ++i) {
            float4 v = zr[i];
            acc[i].x = fmaf(a, v.x, acc[i].x);
            acc[i].y = fmaf(a, v.y, acc[i].y);
            acc[i].z = fmaf(a, v.z, acc[i].z);
            acc[i].w = fmaf(a, v.w, acc[i].w);
        }
    }

    float s = 0.f, ss = 0.f;
#pragma unroll
    for (int i = 0; i < 16; ++i) {
        s  += acc[i].x + acc[i].y + acc[i].z + acc[i].w;
        ss += acc[i].x * acc[i].x + acc[i].y * acc[i].y +
              acc[i].z * acc[i].z + acc[i].w * acc[i].w;
    }
    sred[dg * 32 + j]  = s;
    ssred[dg * 32 + j] = ss;
    __syncthreads();
    float S = 0.f, SS = 0.f;
#pragma unroll
    for (int i = 0; i < 8; ++i) { S += sred[i * 32 + j]; SS += ssred[i * 32 + j]; }
    const float mean = S * (1.0f / (float)DD);
    const float var  = SS * (1.0f / (float)DD) - mean * mean;
    const float rs   = rsqrtf(var + 1e-5f);

    const int e = sidx[j];
    _Float16* outp = Xh + ((size_t)e * BB + b) * DD + d0;
    const float4* g4 = (const float4*)(gamma + d0);
    const float4* t4 = (const float4*)(beta + d0);
#pragma unroll
    for (int i = 0; i < 16; ++i) {
        float4 g = g4[i], t = t4[i], a = acc[i];
        h4 o;
        o[0] = (_Float16)((a.x - mean) * rs * g.x + t.x);
        o[1] = (_Float16)((a.y - mean) * rs * g.y + t.y);
        o[2] = (_Float16)((a.z - mean) * rs * g.z + t.z);
        o[3] = (_Float16)((a.w - mean) * rs * g.w + t.w);
        *(h4*)(outp + i * 4) = o;
    }
}

// ---------------------------------------------------------------------------
// Kernel 2: fused per-expert MLP with f16 WMMA 16x16x32 (f32 accum).
// grid = 32 experts * 16 row-tiles of 64 rows; 256 threads = 8 waves.
// Each wave owns one N-tile and all 4 M-tiles -> each B load reused 4x.
// ---------------------------------------------------------------------------
__global__ __launch_bounds__(256) void k2_expert_mlp(
    const _Float16* __restrict__ Xh, const int* __restrict__ nsrc,
    const _Float16* __restrict__ W1t, const float* __restrict__ b1,
    const _Float16* __restrict__ W2t, const float* __restrict__ b2,
    float* __restrict__ out)
{
    extern __shared__ char dynsmem[];
    _Float16* Xl = (_Float16*)dynsmem;   // 64 x 520 halfs
    _Float16* Hl = Xl + 64 * XS2;        // 64 x 1032 halfs

    const int e   = blockIdx.x >> 4;
    const int r0  = (blockIdx.x & 15) * 64;
    const int tid = threadIdx.x;
    const int lane = tid & 31;
    const int w    = tid >> 5;
    const int nl   = lane & 15;           // A row m / B,C,D col n
    const int kc   = (lane >> 4) << 3;    // A half-chunk offset: 0 or 8
    const int bkc  = (lane >> 4) << 3;    // B k offset: 0 or 8
    const int moff = (lane >> 4) << 3;    // C/D row offset: 0 or 8

    // stage X tile (64 x 512 halfs) into LDS
    {
        const _Float16* xsrc = Xh + ((size_t)e * BB + r0) * DD;
        for (int u = tid; u < 64 * 64; u += 256) {
            int row = u >> 6, c = u & 63;
            *(v8h*)(Xl + row * XS2 + c * 8) = *(const v8h*)(xsrc + row * 512 + c * 8);
        }
    }
    __syncthreads();

    // ---- phase 1: H = elu(X @ W1 + b1), H kept in LDS as f16 ----
    const float* b1p = b1 + e * HH;
    const _Float16* w1b = W1t + (size_t)e * HH * DD + bkc;  // [e][h][d] N-major
#pragma unroll 1
    for (int nt = w; nt < 64; nt += 8) {
        const int nb = nt << 4;
        const _Float16* bp = w1b + (size_t)(nb + nl) * DD;
        v8f acc[4] = {};
#pragma unroll 2
        for (int k0 = 0; k0 < DD; k0 += 32) {
            v8h blo = *(const v8h*)(bp + k0);
            v8h bhi = *(const v8h*)(bp + k0 + 16);
            v16h bv = __builtin_shufflevector(blo, bhi,
                0,1,2,3,4,5,6,7,8,9,10,11,12,13,14,15);
#pragma unroll
            for (int mt = 0; mt < 4; ++mt) {
                const _Float16* ap = Xl + (mt * 16 + nl) * XS2 + k0 + kc;
                v8h alo = *(const v8h*)(ap);
                v8h ahi = *(const v8h*)(ap + 16);
                v16h av = __builtin_shufflevector(alo, ahi,
                    0,1,2,3,4,5,6,7,8,9,10,11,12,13,14,15);
                acc[mt] = __builtin_amdgcn_wmma_f32_16x16x32_f16(
                    false, av, false, bv, (short)0, acc[mt], false, false);
            }
        }
        const float bias = b1p[nb + nl];
#pragma unroll
        for (int mt = 0; mt < 4; ++mt) {
#pragma unroll
            for (int v = 0; v < 8; ++v) {
                float x = acc[mt][v] + bias;
                float h = x > 0.0f ? x : (__expf(x) - 1.0f);  // elu, alpha=1
                Hl[(mt * 16 + v + moff) * HS2 + nb + nl] = (_Float16)h;
            }
        }
    }
    __syncthreads();

    // ---- phase 2: Y = H @ W2 + b2 + residual(X); scatter to out ----
    const float* b2p = b2 + e * DD;
    const _Float16* w2b = W2t + (size_t)e * DD * HH + bkc;  // [e][d][h] N-major
#pragma unroll 1
    for (int nt = w; nt < 32; nt += 8) {
        const int nb = nt << 4;
        const _Float16* bp = w2b + (size_t)(nb + nl) * HH;
        v8f acc[4] = {};
#pragma unroll 2
        for (int k0 = 0; k0 < HH; k0 += 32) {
            v8h blo = *(const v8h*)(bp + k0);
            v8h bhi = *(const v8h*)(bp + k0 + 16);
            v16h bv = __builtin_shufflevector(blo, bhi,
                0,1,2,3,4,5,6,7,8,9,10,11,12,13,14,15);
#pragma unroll
            for (int mt = 0; mt < 4; ++mt) {
                const _Float16* ap = Hl + (mt * 16 + nl) * HS2 + k0 + kc;
                v8h alo = *(const v8h*)(ap);
                v8h ahi = *(const v8h*)(ap + 16);
                v16h av = __builtin_shufflevector(alo, ahi,
                    0,1,2,3,4,5,6,7,8,9,10,11,12,13,14,15);
                acc[mt] = __builtin_amdgcn_wmma_f32_16x16x32_f16(
                    false, av, false, bv, (short)0, acc[mt], false, false);
            }
        }
        const int d = nb + nl;
        const float bias = b2p[d];
#pragma unroll
        for (int mt = 0; mt < 4; ++mt) {
#pragma unroll
            for (int v = 0; v < 8; ++v) {
                const int mrow = mt * 16 + v + moff;
                const int rr   = r0 + mrow;                 // batch index b
                float res = (float)Xl[mrow * XS2 + d];      // residual = Az_ln
                float y = acc[mt][v] + bias + res;
                int nout = nsrc[e * BB + rr];
                out[((size_t)rr * NN + nout) * DD + d] = y;
            }
        }
    }
}

// ---------------------------------------------------------------------------
extern "C" void kernel_launch(void* const* d_in, const int* in_sizes, int n_in,
                              void* d_out, int out_size, void* d_ws, size_t ws_size,
                              hipStream_t stream)
{
    const float* z     = (const float*)d_in[0];
    const int*   idx   = (const int*)  d_in[1];
    const float* A     = (const float*)d_in[2];
    const float* W1    = (const float*)d_in[3];
    const float* b1    = (const float*)d_in[4];
    const float* W2    = (const float*)d_in[5];
    const float* b2    = (const float*)d_in[6];
    const float* gamma = (const float*)d_in[7];
    const float* beta  = (const float*)d_in[8];
    float* out = (float*)d_out;

    // workspace layout (f16 buffers):
    //   Xh  : 32*1024*512 halfs = 32 MB
    //   W1t : 32*1024*512 halfs = 32 MB   ([e][h][d])
    //   W2t : 32*512*1024 halfs = 32 MB   ([e][d][h])
    //   nsrc: 32*1024 i32
    const size_t elems = (size_t)NN * BB * DD;   // == NN*DD*HH per weight too
    _Float16* Xh  = (_Float16*)d_ws;
    _Float16* W1t = Xh  + elems;
    _Float16* W2t = W1t + (size_t)NN * DD * HH;
    int*      nsrc = (int*)(W2t + (size_t)NN * DD * HH);

    // k0: weight convert + transpose (once per launch; graph-capture safe)
    k0_transpose_f16<<<dim3(HH / 32, DD / 32, NN), 256, 0, stream>>>(W1, W1t, DD, HH);
    k0_transpose_f16<<<dim3(DD / 32, HH / 32, NN), 256, 0, stream>>>(W2, W2t, HH, DD);

    const size_t sh1 = (size_t)(32 * ZS + 32 * 33 + 256 + 256) * sizeof(float);
    k1_az_ln_scatter<<<BB, 256, sh1, stream>>>(z, idx, A, gamma, beta, Xh, nsrc);

    const size_t sh2 = (size_t)(64 * XS2 + 64 * HS2) * sizeof(_Float16);
    k2_expert_mlp<<<NN * 16, 256, sh2, stream>>>(Xh, nsrc, W1t, b1, W2t, b2, out);
}
